// GAT_23699629539718
// MI455X (gfx1250) — matrix operand order repaired and checked by reading, not compiled, not run on hardware
//
#include <hip/hip_runtime.h>
#include <hip/hip_bf16.h>

typedef __attribute__((ext_vector_type(16))) _Float16 v16h;
typedef __attribute__((ext_vector_type(8)))  float    v8f;

#define NEG_SLOPE 0.2f
#define F_IN  512
#define HID   8
#define HEADS 8
#define C1    64    // HEADS*HID
#define NCLS  16

// ---------- helpers ----------
static __device__ __forceinline__ unsigned enc_f32(float f) {
    unsigned u = __float_as_uint(f);
    return (u & 0x80000000u) ? ~u : (u | 0x80000000u);   // order-preserving map
}
static __device__ __forceinline__ float dec_f32(unsigned u) {
    u = (u & 0x80000000u) ? (u ^ 0x80000000u) : ~u;
    return __uint_as_float(u);
}
static __device__ __forceinline__ float leaky(float x) {
    return x > 0.f ? x : NEG_SLOPE * x;
}
// K index of element e of a v16h A-fragment (16-bit A 16x32 layout, ISA 7.12.2)
static __device__ __forceinline__ int a_k(int e, int half) {
    int vg = e >> 1;
    int base = (vg < 4) ? (vg * 2) : (16 + (vg - 4) * 2);
    return base + half * 8 + (e & 1);
}

// ---------- weight conversion + fragment swizzle ----------
// W1f layout: [chunk(16)][coltile(4)][lane(32)][e(16)]  (32768 halves)
// W2f layout: [chunk(2)][lane(32)][e(16)]               (1024 halves)
__global__ __launch_bounds__(256) void cvt_weights(const float* __restrict__ W1,
                                                   const float* __restrict__ W2,
                                                   _Float16* __restrict__ W1f,
                                                   _Float16* __restrict__ W2f) {
    int i = blockIdx.x * 256 + threadIdx.x;
    if (i < F_IN * C1) {
        int e = i & 15, lane = (i >> 4) & 31, coltile = (i >> 9) & 3, chunk = i >> 11;
        int n = lane & 15, half = lane >> 4;
        int k = chunk * 32 + (e >> 3) * 16 + half * 8 + (e & 7);
        W1f[i] = (_Float16)W1[k * C1 + coltile * 16 + n];
    }
    if (i < C1 * NCLS) {
        int e = i & 15, lane = (i >> 4) & 31, chunk = i >> 9;
        int n = lane & 15, half = lane >> 4;
        int k = chunk * 32 + (e >> 3) * 16 + half * 8 + (e & 7);
        W2f[i] = (_Float16)W2[k * NCLS + n];
    }
}

// ---------- GEMM1: h1[N,64] = x[N,512] @ W1[512,64] (WMMA f16->f32) ----------
// 32 rows per block; 4 waves each own a 16-col tile over 2 row-tiles.
__global__ __launch_bounds__(128) void gemm1_wmma(const float* __restrict__ x,
                                                  const _Float16* __restrict__ W1f,
                                                  float* __restrict__ h1, int N) {
    // sA layout: [chunk(16)][rowtile(2)][lane(32)][e(16)] = 16384 halves (32 KB)
    __shared__ _Float16 sA[16 * 2 * 32 * 16];
    const int row0 = blockIdx.x * 32;
    const int tid  = threadIdx.x;

    for (int i = tid; i < 16384; i += 128) {
        int e = i & 15, lane = (i >> 4) & 31, rt = (i >> 9) & 1, chunk = i >> 10;
        int m = rt * 16 + (lane & 15), half = lane >> 4;
        int r = row0 + m; if (r >= N) r = N - 1;
        int k = chunk * 32 + a_k(e, half);
        sA[i] = (_Float16)x[(size_t)r * F_IN + k];
    }
    __syncthreads();

    const int wave = tid >> 5, lane = tid & 31;
    const _Float16* wbase = W1f + wave * 512 + lane * 16;
    const _Float16* abase = sA + lane * 16;
    v8f c0 = {}, c1 = {};
#pragma unroll 4
    for (int chunk = 0; chunk < 16; ++chunk) {
        v16h b  = *(const v16h*)(wbase + chunk * 2048);           // 2x global_load_b128
        v16h a0 = *(const v16h*)(abase + chunk * 1024);           // 2x ds_load_b128
        v16h a1 = *(const v16h*)(abase + chunk * 1024 + 512);
        c0 = __builtin_amdgcn_wmma_f32_16x16x32_f16(false, a0, false, b,
                                                    (short)0, c0, false, false);
        c1 = __builtin_amdgcn_wmma_f32_16x16x32_f16(false, a1, false, b,
                                                    (short)0, c1, false, false);
    }

    const int half = lane >> 4, mn = lane & 15, colbase = wave * 16;
    if (row0 + 32 <= N) {                       // block-uniform fast path
#pragma unroll
        for (int r = 0; r < 8; ++r) {
            int M = r + half * 8;
            h1[(size_t)(row0 + M) * C1 + colbase + mn]      = c0[r];
            h1[(size_t)(row0 + 16 + M) * C1 + colbase + mn] = c1[r];
        }
    } else {                                    // rare tail
#pragma unroll
        for (int r = 0; r < 8; ++r) {
            int M = r + half * 8;
            if (row0 + M < N)      h1[(size_t)(row0 + M) * C1 + colbase + mn]      = c0[r];
            if (row0 + 16 + M < N) h1[(size_t)(row0 + 16 + M) * C1 + colbase + mn] = c1[r];
        }
    }
}

// ---------- attention coefficients, layer 1 ----------
__global__ __launch_bounds__(256) void attn1(const float* __restrict__ h1,
                                             const float* __restrict__ asw,
                                             const float* __restrict__ adw,
                                             float* __restrict__ as1,
                                             float* __restrict__ ad1, int N) {
    long long t = (long long)blockIdx.x * 256 + threadIdx.x;
    if (t >= (long long)N * HEADS) return;
    int h = (int)(t & 7); long long n = t >> 3;
    const float* hp = h1 + n * C1 + h * HID;
    float s = 0.f, d = 0.f;
#pragma unroll
    for (int c = 0; c < HID; ++c) {
        s += hp[c] * asw[h * HID + c];
        d += hp[c] * adw[h * HID + c];
    }
    as1[t] = s; ad1[t] = d;
}

// ---------- edge pass A: segment max via ordered-uint atomicMax ----------
__global__ __launch_bounds__(256) void edge_max1(const int* __restrict__ ei,
                                                 const float* __restrict__ as1,
                                                 const float* __restrict__ ad1,
                                                 unsigned* __restrict__ emax1,
                                                 int E, int N) {
    long long i = (long long)blockIdx.x * 256 + threadIdx.x;
    if (i >= (long long)E + N) return;
    int s, d;
    if (i < E) { s = ei[i]; d = ei[(long long)E + i]; } else { s = d = (int)(i - E); }
#pragma unroll
    for (int h = 0; h < HEADS; ++h) {
        float v = leaky(as1[(size_t)s * HEADS + h] + ad1[(size_t)d * HEADS + h]);
        atomicMax(&emax1[(size_t)d * HEADS + h], enc_f32(v));
    }
}

// ---------- edge pass B: fused exp / denom / weighted message scatter ----------
__global__ __launch_bounds__(256) void edge_acc1(const int* __restrict__ ei,
                                                 const float* __restrict__ as1,
                                                 const float* __restrict__ ad1,
                                                 const unsigned* __restrict__ emax1,
                                                 const float* __restrict__ h1,
                                                 float* __restrict__ denom1,
                                                 float* __restrict__ acc1,
                                                 int E, int N) {
    long long t = (long long)blockIdx.x * 256 + threadIdx.x;
    if (t >= ((long long)E + N) * HEADS) return;
    long long i = t >> 3; int h = (int)(t & 7);
    int s, d;
    if (i < E) { s = ei[i]; d = ei[(long long)E + i]; } else { s = d = (int)(i - E); }
    float v = leaky(as1[(size_t)s * HEADS + h] + ad1[(size_t)d * HEADS + h]);
    float p = __expf(v - dec_f32(emax1[(size_t)d * HEADS + h]));
    atomicAdd(&denom1[(size_t)d * HEADS + h], p);
    const float* hs = h1 + (size_t)s * C1 + h * HID;
    float* ac = acc1 + (size_t)d * C1 + h * HID;
    __builtin_prefetch(hs, 0, 0);
#pragma unroll
    for (int c = 0; c < HID; ++c) atomicAdd(&ac[c], p * hs[c]);
}

// ---------- finalize layer 1: normalize + bias + ELU, emit fragment-ordered f16 ----
// zf layout per 16-row tile: [tile][chunk(2)][lane(32)][e(16)]
__global__ __launch_bounds__(256) void finalize1(const float* __restrict__ acc1,
                                                 const float* __restrict__ denom1,
                                                 const float* __restrict__ b1,
                                                 _Float16* __restrict__ zf, int N) {
    long long t = (long long)blockIdx.x * 256 + threadIdx.x;
    if (t >= (long long)N * C1) return;
    int c = (int)(t & 63); long long n = t >> 6;
    int h = c >> 3;
    float v = acc1[t] / denom1[n * HEADS + h] + b1[c];
    v = v > 0.f ? v : expm1f(v);                // ELU
    // inverse A-fragment mapping: k=c -> (chunk, half, e)
    int chunk = c >> 5, kk = c & 31;
    int kk16 = kk & 15, half = kk16 >> 3, r = kk16 & 7;
    int vg = (r >> 1) + ((kk >= 16) ? 4 : 0);
    int e = vg * 2 + (r & 1);
    long long tile = n >> 4; int mn = (int)(n & 15);
    zf[tile * 1024 + chunk * 512 + (half * 16 + mn) * 16 + e] = (_Float16)v;
}

// ---------- GEMM2: h2[N,16] = z1[N,64] @ W2[64,16] (WMMA) ----------
__global__ __launch_bounds__(128) void gemm2_wmma(const _Float16* __restrict__ zf,
                                                  const _Float16* __restrict__ W2f,
                                                  float* __restrict__ h2, int N) {
    const int wave = threadIdx.x >> 5, lane = threadIdx.x & 31;
    const int tile = blockIdx.x * 4 + wave;
    const int row0 = tile * 16;
    if (row0 >= N) return;                      // whole wave exits together
    const _Float16* ab = zf + (size_t)tile * 1024 + lane * 16;
    const _Float16* bb = W2f + lane * 16;
    v8f c = {};
#pragma unroll
    for (int chunk = 0; chunk < 2; ++chunk) {
        v16h a = *(const v16h*)(ab + chunk * 512);
        v16h b = *(const v16h*)(bb + chunk * 512);
        c = __builtin_amdgcn_wmma_f32_16x16x32_f16(false, a, false, b,
                                                   (short)0, c, false, false);
    }
    const int half = lane >> 4, mn = lane & 15;
    if (row0 + 16 <= N) {
#pragma unroll
        for (int r = 0; r < 8; ++r)
            h2[(size_t)(row0 + r + half * 8) * NCLS + mn] = c[r];
    } else {
#pragma unroll
        for (int r = 0; r < 8; ++r) {
            int row = row0 + r + half * 8;
            if (row < N) h2[(size_t)row * NCLS + mn] = c[r];
        }
    }
}

// ---------- layer 2 attention / edges / finalize (H=1, C=16) ----------
__global__ __launch_bounds__(256) void attn2(const float* __restrict__ h2,
                                             const float* __restrict__ asw,
                                             const float* __restrict__ adw,
                                             float* __restrict__ as2,
                                             float* __restrict__ ad2, int N) {
    long long n = (long long)blockIdx.x * 256 + threadIdx.x;
    if (n >= N) return;
    const float* hp = h2 + n * NCLS;
    float s = 0.f, d = 0.f;
#pragma unroll
    for (int c = 0; c < NCLS; ++c) { s += hp[c] * asw[c]; d += hp[c] * adw[c]; }
    as2[n] = s; ad2[n] = d;
}

__global__ __launch_bounds__(256) void edge_max2(const int* __restrict__ ei,
                                                 const float* __restrict__ as2,
                                                 const float* __restrict__ ad2,
                                                 unsigned* __restrict__ emax2,
                                                 int E, int N) {
    long long i = (long long)blockIdx.x * 256 + threadIdx.x;
    if (i >= (long long)E + N) return;
    int s, d;
    if (i < E) { s = ei[i]; d = ei[(long long)E + i]; } else { s = d = (int)(i - E); }
    float v = leaky(as2[s] + ad2[d]);
    atomicMax(&emax2[d], enc_f32(v));
}

__global__ __launch_bounds__(256) void edge_acc2(const int* __restrict__ ei,
                                                 const float* __restrict__ as2,
                                                 const float* __restrict__ ad2,
                                                 const unsigned* __restrict__ emax2,
                                                 const float* __restrict__ h2,
                                                 float* __restrict__ denom2,
                                                 float* __restrict__ acc2,
                                                 int E, int N) {
    long long i = (long long)blockIdx.x * 256 + threadIdx.x;
    if (i >= (long long)E + N) return;
    int s, d;
    if (i < E) { s = ei[i]; d = ei[(long long)E + i]; } else { s = d = (int)(i - E); }
    float v = leaky(as2[s] + ad2[d]);
    float p = __expf(v - dec_f32(emax2[d]));
    atomicAdd(&denom2[d], p);
    const float* hs = h2 + (size_t)s * NCLS;
    float* ac = acc2 + (size_t)d * NCLS;
#pragma unroll
    for (int c = 0; c < NCLS; ++c) atomicAdd(&ac[c], p * hs[c]);
}

__global__ __launch_bounds__(256) void finalize2(const float* __restrict__ acc2,
                                                 const float* __restrict__ denom2,
                                                 const float* __restrict__ b2,
                                                 float* __restrict__ out, int N) {
    long long t = (long long)blockIdx.x * 256 + threadIdx.x;
    if (t >= (long long)N * NCLS) return;
    out[t] = acc2[t] / denom2[t >> 4] + b2[t & 15];
}

// ---------- host ----------
extern "C" void kernel_launch(void* const* d_in, const int* in_sizes, int n_in,
                              void* d_out, int out_size, void* d_ws, size_t ws_size,
                              hipStream_t stream) {
    const float* x        = (const float*)d_in[0];
    const int*   ei       = (const int*)  d_in[1];   // JAX x64 off -> int32
    const float* W1       = (const float*)d_in[2];
    const float* att_src1 = (const float*)d_in[3];
    const float* att_dst1 = (const float*)d_in[4];
    const float* b1       = (const float*)d_in[5];
    const float* W2       = (const float*)d_in[6];
    const float* att_src2 = (const float*)d_in[7];
    const float* att_dst2 = (const float*)d_in[8];
    const float* b2       = (const float*)d_in[9];
    float*       out      = (float*)d_out;

    const int N = in_sizes[0] / F_IN;
    const int E = in_sizes[1] / 2;
    const long long TE = (long long)E + N;
    const int tiles = (N + 15) / 16;

    char* ws = (char*)d_ws;
    size_t off = 0;
    auto carve = [&](size_t bytes) -> void* {
        void* p = ws + off;
        off = (off + bytes + 255) & ~(size_t)255;
        return p;
    };
    float*     h1    = (float*)    carve((size_t)N * C1 * 4);
    float*     as1   = (float*)    carve((size_t)N * HEADS * 4);
    float*     ad1   = (float*)    carve((size_t)N * HEADS * 4);
    _Float16*  zf    = (_Float16*) carve((size_t)tiles * 1024 * 2);
    float*     h2    = (float*)    carve((size_t)N * NCLS * 4);
    float*     as2   = (float*)    carve((size_t)N * 4);
    float*     ad2   = (float*)    carve((size_t)N * 4);
    _Float16*  W1f   = (_Float16*) carve((size_t)F_IN * C1 * 2);
    _Float16*  W2f   = (_Float16*) carve((size_t)C1 * NCLS * 2);
    // contiguous zero-init region (re-zeroed every call; graph-capture safe)
    size_t zoff = off;
    unsigned*  emax1  = (unsigned*) carve((size_t)N * HEADS * 4);
    float*     denom1 = (float*)    carve((size_t)N * HEADS * 4);
    float*     acc1   = (float*)    carve((size_t)N * C1 * 4);
    unsigned*  emax2  = (unsigned*) carve((size_t)N * 4);
    float*     denom2 = (float*)    carve((size_t)N * 4);
    float*     acc2   = (float*)    carve((size_t)N * NCLS * 4);
    size_t zbytes = off - zoff;

    hipMemsetAsync(ws + zoff, 0, zbytes, stream);

    cvt_weights<<<(F_IN * C1 + 255) / 256, 256, 0, stream>>>(W1, W2, W1f, W2f);

    // ---- layer 1 ----
    gemm1_wmma<<<(N + 31) / 32, 128, 0, stream>>>(x, W1f, h1, N);
    attn1<<<(unsigned)(((long long)N * HEADS + 255) / 256), 256, 0, stream>>>(
        h1, att_src1, att_dst1, as1, ad1, N);
    edge_max1<<<(unsigned)((TE + 255) / 256), 256, 0, stream>>>(ei, as1, ad1, emax1, E, N);
    edge_acc1<<<(unsigned)((TE * HEADS + 255) / 256), 256, 0, stream>>>(
        ei, as1, ad1, emax1, h1, denom1, acc1, E, N);
    finalize1<<<(unsigned)(((long long)N * C1 + 255) / 256), 256, 0, stream>>>(
        acc1, denom1, b1, zf, N);

    // ---- layer 2 ----
    gemm2_wmma<<<(tiles + 3) / 4, 128, 0, stream>>>(zf, W2f, h2, N);
    attn2<<<(N + 255) / 256, 256, 0, stream>>>(h2, att_src2, att_dst2, as2, ad2, N);
    edge_max2<<<(unsigned)((TE + 255) / 256), 256, 0, stream>>>(ei, as2, ad2, emax2, E, N);
    edge_acc2<<<(unsigned)((TE + 255) / 256), 256, 0, stream>>>(
        ei, as2, ad2, emax2, h2, denom2, acc2, E, N);
    finalize2<<<(unsigned)(((long long)N * NCLS + 255) / 256), 256, 0, stream>>>(
        acc2, denom2, b2, out, N);
}
